// GINConv_68539088109729
// MI455X (gfx1250) — matrix operand order repaired and checked by reading, not compiled, not run on hardware
//
#include <hip/hip_runtime.h>
#include <hip/hip_bf16.h>

// ---------------------------------------------------------------------------
// GIN layer for MI455X (gfx1250, wave32, WMMA).
//   agg = segment_sum(x[src], dst); h = (1+eps)x + agg
//   out = GELU(LN(h@W1+b1)) @ W2 + b2
// ---------------------------------------------------------------------------

typedef __attribute__((ext_vector_type(16))) __bf16 v16bf;
typedef __attribute__((ext_vector_type(8)))  float  v8f;

#define IN_DIMK 128
#define HIDK    512
#define EMBK    128
#define LN_EPSF 1e-5f

__device__ __forceinline__ unsigned short f2bf(float f) {
  unsigned u = __builtin_bit_cast(unsigned, f);
  u += 0x7FFFu + ((u >> 16) & 1u);            // round-to-nearest-even
  return (unsigned short)(u >> 16);
}
__device__ __forceinline__ __bf16 bfrom(unsigned short u) {
  return __builtin_bit_cast(__bf16, u);
}

// ---------------- kernel 1: H = (1+eps) * x  (re-inits workspace) ----------
__global__ void k_init(const float* __restrict__ x, const float* __restrict__ epsp,
                       float* __restrict__ H, size_t nelem4) {
  size_t i = (size_t)blockIdx.x * blockDim.x + threadIdx.x;
  if (i >= nelem4) return;
  float s = 1.0f + epsp[0];
  float4 v = reinterpret_cast<const float4*>(x)[i];
  v.x *= s; v.y *= s; v.z *= s; v.w *= s;
  reinterpret_cast<float4*>(H)[i] = v;
}

// ---------------- kernel 2: weights -> bf16, K-contiguous transposed -------
__global__ void k_prep(const float* __restrict__ W1, const float* __restrict__ W2,
                       unsigned short* __restrict__ w1t, unsigned short* __restrict__ w2t) {
  int i = blockIdx.x * blockDim.x + threadIdx.x;
  if (i < IN_DIMK * HIDK) {            // W1[k][c] -> w1t[c*128 + k]
    int k = i / HIDK, c = i % HIDK;
    w1t[c * IN_DIMK + k] = f2bf(W1[i]);
  }
  if (i < HIDK * EMBK) {               // W2[k][c] -> w2t[c*512 + k]
    int k = i / EMBK, c = i % EMBK;
    w2t[c * HIDK + k] = f2bf(W2[i]);
  }
}

// ---------------- kernel 3: edge scatter-add (L2-resident f32 atomics) -----
__global__ void k_scatter(const float* __restrict__ x, const long long* __restrict__ ei,
                          float* __restrict__ H, int E) {
  size_t gid = (size_t)blockIdx.x * blockDim.x + threadIdx.x;
  int e    = (int)(gid >> 5);
  int lane = (int)(gid & 31);
  if (e >= E) return;
  int s = (int)ei[e];
  int d = (int)ei[(size_t)E + e];
  float4 v = *reinterpret_cast<const float4*>(x + (size_t)s * IN_DIMK + lane * 4);
  float* dst = H + (size_t)d * IN_DIMK + lane * 4;
  unsafeAtomicAdd(dst + 0, v.x);
  unsafeAtomicAdd(dst + 1, v.y);
  unsafeAtomicAdd(dst + 2, v.z);
  unsafeAtomicAdd(dst + 3, v.w);
}

// ---------------- kernel 4: fused MLP, 16 rows / block, bf16 WMMA ----------
__global__ void __launch_bounds__(256)
k_mlp(const float* __restrict__ H,
      const unsigned short* __restrict__ w1t, const unsigned short* __restrict__ w2t,
      const float* __restrict__ b1, const float* __restrict__ lng,
      const float* __restrict__ lnb, const float* __restrict__ b2,
      float* __restrict__ out, int nodes) {
  __shared__ unsigned short sA [16][IN_DIMK + 8];   // h tile, bf16
  __shared__ float          sHid[16][HIDK + 4];     // hidden, fp32 (for LN)
  __shared__ unsigned short sHb [16][HIDK + 8];     // post-GELU hidden, bf16

  const int tid  = threadIdx.x;
  const int wave = tid >> 5;
  const int lane = tid & 31;
  const int lr   = lane & 15;        // column-within-tile / row selector
  const bool hi  = lane >= 16;
  const int base = blockIdx.x * 16;

  // hint far B-operand of GEMM2 into cache early (global_prefetch_b8)
  __builtin_prefetch(w2t + (size_t)(wave * 16 + lr) * HIDK, 0, 1);

  // ---- phase 0: stage 16x128 h tile as bf16 in LDS ----
  for (int i = tid; i < 16 * IN_DIMK; i += 256) {
    int r = i / IN_DIMK, c = i % IN_DIMK;
    float v = (base + r < nodes) ? H[(size_t)(base + r) * IN_DIMK + c] : 0.0f;
    sA[r][c] = f2bf(v);
  }
  __syncthreads();

  // ---- phase 1: GEMM1 (16x128 @ 128x512), 4 col-tiles per wave ----
  for (int t = 0; t < 4; ++t) {
    const int col0 = (wave * 4 + t) * 16;
    v8f acc = {};
#pragma unroll
    for (int kt = 0; kt < 4; ++kt) {
      const int kb = kt * 32;
      const int ka = kb + (hi ? 16 : 0);
      v16bf a, b;
#pragma unroll
      for (int j = 0; j < 8; ++j) {
        a[j]     = bfrom(sA[lr][ka + j]);        // K = ka..ka+7
        a[8 + j] = bfrom(sA[lr][ka + 8 + j]);    // K = ka+8..ka+15
      }
      const unsigned short* bp = w1t + (size_t)(col0 + lr) * IN_DIMK + kb + (hi ? 16 : 0);
#pragma unroll
      for (int j = 0; j < 16; ++j) b[j] = bfrom(bp[j]);
      acc = __builtin_amdgcn_wmma_f32_16x16x32_bf16(false, a, false, b,
                                                    (short)0, acc, false, false);
    }
#pragma unroll
    for (int i = 0; i < 8; ++i) {                // D layout: M = i + (hi?8:0)
      int r = i + (hi ? 8 : 0);
      int c = col0 + lr;
      sHid[r][c] = acc[i] + b1[c];
    }
  }
  __syncthreads();

  // ---- phase 2: LayerNorm + exact-erf GELU (2 rows / wave) ----
  for (int r = wave; r < 16; r += 8) {
    float s = 0.0f, s2 = 0.0f;
    for (int c = lane; c < HIDK; c += 32) {
      float v = sHid[r][c];
      s += v; s2 += v * v;
    }
#pragma unroll
    for (int off = 16; off >= 1; off >>= 1) {    // wave32 tree reduce
      s  += __shfl_xor(s,  off, 32);
      s2 += __shfl_xor(s2, off, 32);
    }
    float mu  = s * (1.0f / HIDK);
    float var = s2 * (1.0f / HIDK) - mu * mu;
    float rs  = rsqrtf(var + LN_EPSF);
    for (int c = lane; c < HIDK; c += 32) {
      float v = (sHid[r][c] - mu) * rs * lng[c] + lnb[c];
      float g = 0.5f * v * (1.0f + erff(v * 0.70710678118654752f));
      sHb[r][c] = f2bf(g);
    }
  }
  __syncthreads();

  // ---- phase 3: GEMM2 (16x512 @ 512x128), 1 col-tile per wave ----
  {
    const int col0 = wave * 16;
    v8f acc = {};
#pragma unroll
    for (int kt = 0; kt < 16; ++kt) {
      const int kb = kt * 32;
      const int ka = kb + (hi ? 16 : 0);
      v16bf a, b;
#pragma unroll
      for (int j = 0; j < 8; ++j) {
        a[j]     = bfrom(sHb[lr][ka + j]);
        a[8 + j] = bfrom(sHb[lr][ka + 8 + j]);
      }
      const unsigned short* bp = w2t + (size_t)(col0 + lr) * HIDK + kb + (hi ? 16 : 0);
#pragma unroll
      for (int j = 0; j < 16; ++j) b[j] = bfrom(bp[j]);
      acc = __builtin_amdgcn_wmma_f32_16x16x32_bf16(false, a, false, b,
                                                    (short)0, acc, false, false);
    }
#pragma unroll
    for (int i = 0; i < 8; ++i) {
      int r = base + i + (hi ? 8 : 0);
      int c = col0 + lr;
      if (r < nodes) out[(size_t)r * EMBK + c] = acc[i] + b2[c];
    }
  }
}

// ---------------------------------------------------------------------------
extern "C" void kernel_launch(void* const* d_in, const int* in_sizes, int n_in,
                              void* d_out, int out_size, void* d_ws, size_t ws_size,
                              hipStream_t stream) {
  const float*     x    = (const float*)d_in[0];
  const long long* ei   = (const long long*)d_in[1];
  const float*     epsp = (const float*)d_in[2];
  const float*     W1   = (const float*)d_in[3];
  const float*     b1   = (const float*)d_in[4];
  const float*     lng  = (const float*)d_in[5];
  const float*     lnb  = (const float*)d_in[6];
  const float*     W2   = (const float*)d_in[7];
  const float*     b2   = (const float*)d_in[8];
  float*           out  = (float*)d_out;

  const int nodes = in_sizes[0] / IN_DIMK;
  const int E     = in_sizes[1] / 2;
  const size_t nelem = (size_t)nodes * IN_DIMK;

  // workspace layout: H (fp32 N*128) | W1^T bf16 | W2^T bf16
  float* H = (float*)d_ws;
  unsigned short* w1t = (unsigned short*)(H + nelem);
  unsigned short* w2t = w1t + (size_t)IN_DIMK * HIDK;

  k_init<<<(unsigned)((nelem / 4 + 255) / 256), 256, 0, stream>>>(x, epsp, H, nelem / 4);
  k_prep<<<(IN_DIMK * HIDK + 255) / 256, 256, 0, stream>>>(W1, W2, w1t, w2t);

  size_t sthreads = (size_t)E * 32;
  k_scatter<<<(unsigned)((sthreads + 255) / 256), 256, 0, stream>>>(x, ei, H, E);

  k_mlp<<<(nodes + 15) / 16, 256, 0, stream>>>(H, w1t, w2t, b1, lng, lnb, b2, out, nodes);
}
